// RewardHead_22411139350741
// MI455X (gfx1250) — matrix operand order repaired and checked by reading, not compiled, not run on hardware
//
#include <hip/hip_runtime.h>
#include <hip/hip_bf16.h>
#include <math.h>

// ---------------------------------------------------------------------------
// RewardHead for MI455X (gfx1250, wave32, WMMA).
// Compute-bound (~12.9 GFLOP vs ~23 MB HBM traffic): all GEMMs mapped to
// v_wmma_f32_16x16x32_bf16. Weights pre-packed to bf16 B-fragment layout in
// workspace (L2-resident; 192MB L2). One wave = one 16-row tile.
// K/V GEMMs are nt-outer with all 3 tokens fused on one B-tile load, cutting
// per-wave L2 weight traffic from 384KB to 256KB (Wk/Wv read once, not 3x).
// tanh via v_exp+v_rcp (~5 VALU ops vs ~30 for libm tanhf).
// ---------------------------------------------------------------------------

typedef __attribute__((ext_vector_type(16))) __bf16 v16bf;
typedef __attribute__((ext_vector_type(8)))  __bf16 v8bf;
typedef __attribute__((ext_vector_type(8)))  float  v8f;

// ---- packed-weight offsets in workspace (bf16 elements, after 256B header) --
#define OFF_ACT_WX   0        // 96x128  (KT=3, NT=8)
#define OFF_CTL_WX   12288    // 128x128 (4,8)
#define OFF_CHT_WX   28672    // 80->96 x128 (3,8)
#define OFF_ACT_WZ   40960    // 32x64   (1,4)
#define OFF_CTL_WZ   43008    // 64x64   (2,4)
#define OFF_CHT_WZ   47104    // 16->32 x64 (1,4)
#define OFF_WE       49152    // 64x128  (2,8)
#define OFF_WQ       57344    // 128x128 (4,8)
#define OFF_WK       73728
#define OFF_WV       90112
#define OFF_WO       106496
#define OFF_FORM     122880   // 128x64 (4,4)  -> end 131072 elems = 256KB

static __device__ __forceinline__ v8f zero8() {
  v8f z = {0.f,0.f,0.f,0.f,0.f,0.f,0.f,0.f};
  return z;
}

// fast tanh: 1 - 2/(exp(2x)+1). Saturates correctly at +/-inf, NaN propagates.
static __device__ __forceinline__ float fast_tanh(float x) {
  float e = __expf(2.0f * x);
  return 1.0f - 2.0f * __builtin_amdgcn_rcpf(e + 1.0f);
}

// A-fragment (16x32 bf16) from a bf16 LDS tile (row-major, stride in elems).
// ISA layout: lanes0-15: M=lane, K = e (e<8) / e+8 (e>=8);
//             lanes16-31: M=lane-16, K = e+8 (e<8) / e+16 (e>=8).
// Per lane: two contiguous 16B chunks -> 2x ds_load_b128, no converts.
static __device__ __forceinline__ v16bf load_A_lds(const __bf16* base, int stride,
                                                   int colbase, int lane) {
  const int m  = lane & 15;
  const int hi = lane >> 4;
  const __bf16* rp = base + m * stride + colbase + 8 * hi;
  v8bf lo = *(const v8bf*)(rp);        // K = 0..7   (+8*hi)
  v8bf hh = *(const v8bf*)(rp + 16);   // K = 16..23 (+8*hi)
  v16bf a;
#pragma unroll
  for (int e = 0; e < 8; ++e) { a[e] = lo[e]; a[e + 8] = hh[e]; }
  return a;
}

// B-fragment (32x16 bf16): pre-packed so each lane reads 16 contiguous bf16.
static __device__ __forceinline__ v16bf load_B(const __bf16* pack, int tile, int lane) {
  return *(const v16bf*)(pack + (size_t)tile * 512 + lane * 16);
}

template <int NT, int NK>
static __device__ __forceinline__ void gemm_acc(const __bf16* A, int strideA,
                                                const int* kb, const __bf16* pack,
                                                v8f acc[NT], int lane) {
#pragma unroll
  for (int kt = 0; kt < NK; ++kt) {
    v16bf a = load_A_lds(A, strideA, kb[kt], lane);
#pragma unroll
    for (int nt = 0; nt < NT; ++nt) {
      v16bf b = load_B(pack, kt * NT + nt, lane);
      acc[nt] = __builtin_amdgcn_wmma_f32_16x16x32_bf16(
          false, a, false, b, (short)0, acc[nt], false, false);
    }
  }
}

// sum across the 16 lanes sharing a D-tile row group (xor stays within halves)
static __device__ __forceinline__ float red16(float v) {
  v += __shfl_xor(v, 1);
  v += __shfl_xor(v, 2);
  v += __shfl_xor(v, 4);
  v += __shfl_xor(v, 8);
  return v;
}

// ---------------------------------------------------------------------------
// Prep kernel 1: pack a f32 [K,N] weight into bf16 WMMA-B tiles (zero K-pad).
// ---------------------------------------------------------------------------
__global__ void pack_kernel(const float* __restrict__ W, __bf16* __restrict__ out,
                            int K, int N, int KT, int NT) {
  int total = KT * NT * 512;
  for (int idx = blockIdx.x * blockDim.x + threadIdx.x; idx < total;
       idx += gridDim.x * blockDim.x) {
    int e    = idx & 15;
    int ln   = (idx >> 4) & 31;
    int tile = idx >> 9;
    int nt = tile % NT, kt = tile / NT;
    int k = kt * 32 + ((e >> 3) << 4) + ((ln >> 4) << 3) + (e & 7);
    int n = nt * 16 + (ln & 15);
    float val = (k < K) ? W[k * N + n] : 0.0f;
    out[idx] = (__bf16)val;
  }
}

// ---------------------------------------------------------------------------
// Prep kernel 2: spectral norms. sigma(form_W[128,64]) via power iteration,
// ||cons_W||_2.  scal[0]=1/(||cons||+1e-6), scal[1]=1/(sigma+1e-6).
// ---------------------------------------------------------------------------
__global__ void spec_kernel(const float* __restrict__ formW,
                            const float* __restrict__ consW,
                            float* __restrict__ scal) {
  __shared__ float v[64], u[128], red[1];
  int t = threadIdx.x;  // 128 threads
  if (t < 64) v[t] = 1.0f;
  __syncthreads();
  float sigma = 0.0f;
  for (int it = 0; it < 30; ++it) {
    if (t == 0) {
      float s = 0.f;
      for (int i = 0; i < 64; ++i) s += v[i] * v[i];
      red[0] = rsqrtf(s + 1e-30f);
    }
    __syncthreads();
    if (t < 64) v[t] *= red[0];
    __syncthreads();
    {   // u = W v  (thread t handles row t)
      float s = 0.f;
      for (int n = 0; n < 64; ++n) s += formW[t * 64 + n] * v[n];
      u[t] = s;
    }
    __syncthreads();
    if (t == 0) {
      float s = 0.f;
      for (int i = 0; i < 128; ++i) s += u[i] * u[i];
      red[0] = sqrtf(s);
    }
    __syncthreads();
    sigma = red[0];
    if (t < 64) {   // v = W^T u
      float s = 0.f;
      for (int k = 0; k < 128; ++k) s += formW[k * 64 + t] * u[k];
      v[t] = s;
    }
    __syncthreads();
  }
  if (t == 0) {
    scal[1] = 1.0f / (sigma + 1e-6f);
    float s = 0.f;
    for (int i = 0; i < 128; ++i) s += consW[i] * consW[i];
    scal[0] = 1.0f / (sqrtf(s) + 1e-6f);
  }
}

// ---------------------------------------------------------------------------
// Main kernel: one wave (32 threads) = one 16-row tile.
// bf16 LDS input layout / row (stride 176): [act 0..31][rw 32..47][z 48..111][ctl 112..175]
// ---------------------------------------------------------------------------
__global__ __launch_bounds__(32)
void reward_head_kernel(const float* __restrict__ z, const float* __restrict__ action,
                        const float* __restrict__ rw, const float* __restrict__ control,
                        const float* __restrict__ act_bx, const float* __restrict__ ctl_bx,
                        const float* __restrict__ cht_bx, const float* __restrict__ be,
                        const float* __restrict__ cons_W, const float* __restrict__ cons_b,
                        const float* __restrict__ form_b,
                        const __bf16* __restrict__ pack, const float* __restrict__ scal,
                        float* __restrict__ out) {
  __shared__ __bf16 sIN[16 * 176];     // bf16 staged inputs (A-fragment source)
  __shared__ __bf16 sX[3 * 16 * 128];  // token activations; sX[0] reused for feat
  __shared__ __bf16 sT[16 * 128];      // x_q, then attn-weighted V
  __shared__ float  sCTL[16 * 64];     // f32 control for the form-head dot
  __shared__ float  sD2[3][16];
  __shared__ float  sScore[3][16];
  __shared__ float  sAttn[3][16];
  __shared__ float  sNC[16];
  __shared__ float  sCons[16];

  const int lane = threadIdx.x;
  const int l15 = lane & 15;
  const int hi  = lane >> 4;
  const int row0 = blockIdx.x * 16;

  // ---- stage inputs (global f32 -> LDS bf16; control also kept f32) --------
  for (int i = lane; i < 16 * 32; i += 32)
    sIN[(i >> 5) * 176 + (i & 31)] = (__bf16)action[(size_t)(row0 + (i >> 5)) * 32 + (i & 31)];
  for (int i = lane; i < 16 * 16; i += 32)
    sIN[(i >> 4) * 176 + 32 + (i & 15)] = (__bf16)rw[(size_t)(row0 + (i >> 4)) * 16 + (i & 15)];
  for (int i = lane; i < 16 * 64; i += 32)
    sIN[(i >> 6) * 176 + 48 + (i & 63)] = (__bf16)z[(size_t)(row0 + (i >> 6)) * 64 + (i & 63)];
  for (int i = lane; i < 16 * 64; i += 32) {
    float c = control[(size_t)(row0 + (i >> 6)) * 64 + (i & 63)];
    sIN[(i >> 6) * 176 + 112 + (i & 63)] = (__bf16)c;
    sCTL[i] = c;
  }
  __syncthreads();

  // ---- tokenizer x = tanh([u,z] @ Wx + bx)  --------------------------------
  {
    const int kb[3] = {0, 48, 80};  // [action | z]
    v8f acc[8]; for (int i = 0; i < 8; ++i) acc[i] = zero8();
    gemm_acc<8, 3>(sIN, 176, kb, pack + OFF_ACT_WX, acc, lane);
#pragma unroll
    for (int nt = 0; nt < 8; ++nt) {
      int col = nt * 16 + l15;
      float b = act_bx[col];
#pragma unroll
      for (int v = 0; v < 8; ++v)
        sX[0 * 2048 + (v + 8 * hi) * 128 + col] = (__bf16)fast_tanh(acc[nt][v] + b);
    }
  }
  {
    const int kb[4] = {112, 144, 48, 80};  // [control | z]
    v8f acc[8]; for (int i = 0; i < 8; ++i) acc[i] = zero8();
    gemm_acc<8, 4>(sIN, 176, kb, pack + OFF_CTL_WX, acc, lane);
#pragma unroll
    for (int nt = 0; nt < 8; ++nt) {
      int col = nt * 16 + l15;
      float b = ctl_bx[col];
#pragma unroll
      for (int v = 0; v < 8; ++v)
        sX[1 * 2048 + (v + 8 * hi) * 128 + col] = (__bf16)fast_tanh(acc[nt][v] + b);
    }
  }
  {
    const int kb[3] = {32, 64, 96};  // [rw | z], K padded 80->96 (zero rows)
    v8f acc[8]; for (int i = 0; i < 8; ++i) acc[i] = zero8();
    gemm_acc<8, 3>(sIN, 176, kb, pack + OFF_CHT_WX, acc, lane);
#pragma unroll
    for (int nt = 0; nt < 8; ++nt) {
      int col = nt * 16 + l15;
      float b = cht_bx[col];
#pragma unroll
      for (int v = 0; v < 8; ++v)
        sX[2 * 2048 + (v + 8 * hi) * 128 + col] = (__bf16)fast_tanh(acc[nt][v] + b);
    }
  }

  // ---- dist2_t = || tanh(u @ Wz) ||^2 per row  (since ctx_z = z + tanh(u@Wz))
#pragma unroll
  for (int t = 0; t < 3; ++t) {
    v8f acc[4]; for (int i = 0; i < 4; ++i) acc[i] = zero8();
    if (t == 0) { const int kb[1] = {0};        gemm_acc<4, 1>(sIN, 176, kb, pack + OFF_ACT_WZ, acc, lane); }
    if (t == 1) { const int kb[2] = {112, 144}; gemm_acc<4, 2>(sIN, 176, kb, pack + OFF_CTL_WZ, acc, lane); }
    if (t == 2) { const int kb[1] = {32};       gemm_acc<4, 1>(sIN, 176, kb, pack + OFF_CHT_WZ, acc, lane); }
    float p[8];
#pragma unroll
    for (int v = 0; v < 8; ++v) p[v] = 0.f;
#pragma unroll
    for (int nt = 0; nt < 4; ++nt)
#pragma unroll
      for (int v = 0; v < 8; ++v) { float g = fast_tanh(acc[nt][v]); p[v] += g * g; }
#pragma unroll
    for (int v = 0; v < 8; ++v) {
      float s = red16(p[v]);
      if (l15 == 0) sD2[t][v + 8 * hi] = s;
    }
  }

  // ---- x_q = z @ We + be ---------------------------------------------------
  {
    const int kb[2] = {48, 80};
    v8f acc[8]; for (int i = 0; i < 8; ++i) acc[i] = zero8();
    gemm_acc<8, 2>(sIN, 176, kb, pack + OFF_WE, acc, lane);
#pragma unroll
    for (int nt = 0; nt < 8; ++nt) {
      int col = nt * 16 + l15;
      float b = be[col];
#pragma unroll
      for (int v = 0; v < 8; ++v)
        sT[(v + 8 * hi) * 128 + col] = (__bf16)(acc[nt][v] + b);
    }
  }
  __syncthreads();

  // ---- q = x_q @ Wq (kept in registers) ------------------------------------
  v8f qacc[8]; for (int i = 0; i < 8; ++i) qacc[i] = zero8();
  {
    const int kb[4] = {0, 32, 64, 96};
    gemm_acc<8, 4>(sT, 128, kb, pack + OFF_WQ, qacc, lane);
  }

  // ---- K GEMMs: nt-outer, 3 tokens fused on one B-tile load ---------------
  // scores[t][row] = sum_col q[row,col]*k_t[row,col]
  {
    float pK[3][8];
#pragma unroll
    for (int t = 0; t < 3; ++t)
#pragma unroll
      for (int v = 0; v < 8; ++v) pK[t][v] = 0.f;

#pragma unroll
    for (int nt = 0; nt < 8; ++nt) {
      v8f kacc[3]; for (int t = 0; t < 3; ++t) kacc[t] = zero8();
#pragma unroll
      for (int kt = 0; kt < 4; ++kt) {
        v16bf b = load_B(pack + OFF_WK, kt * 8 + nt, lane);
#pragma unroll
        for (int t = 0; t < 3; ++t) {
          v16bf a = load_A_lds(sX + t * 2048, 128, kt * 32, lane);
          kacc[t] = __builtin_amdgcn_wmma_f32_16x16x32_bf16(
              false, a, false, b, (short)0, kacc[t], false, false);
        }
      }
#pragma unroll
      for (int t = 0; t < 3; ++t)
#pragma unroll
        for (int v = 0; v < 8; ++v) pK[t][v] += qacc[nt][v] * kacc[t][v];
    }
#pragma unroll
    for (int t = 0; t < 3; ++t)
#pragma unroll
      for (int v = 0; v < 8; ++v) {
        float s = red16(pK[t][v]);
        if (l15 == 0) sScore[t][v + 8 * hi] = s;
      }
  }
  __syncthreads();

  // ---- softmax over the 3 tokens (lanes 0..15, one row each) ---------------
  if (lane < 16) {
    const float invs = 0.08838834764831845f;  // 1/sqrt(128)
    float s0 = sScore[0][lane] * invs - sD2[0][lane];
    float s1 = sScore[1][lane] * invs - sD2[1][lane];
    float s2 = sScore[2][lane] * invs - sD2[2][lane];
    float m = fmaxf(s0, fmaxf(s1, s2));
    float e0 = __expf(s0 - m), e1 = __expf(s1 - m), e2 = __expf(s2 - m);
    float inv = 1.0f / (e0 + e1 + e2);
    sAttn[0][lane] = e0 * inv;
    sAttn[1][lane] = e1 * inv;
    sAttn[2][lane] = e2 * inv;
  }
  __syncthreads();

  // ---- V GEMMs: nt-outer, 3 tokens fused; feat slice finalized per nt ------
  {
    float av[3][8];
#pragma unroll
    for (int t = 0; t < 3; ++t)
#pragma unroll
      for (int v = 0; v < 8; ++v) av[t][v] = sAttn[t][v + 8 * hi];

#pragma unroll
    for (int nt = 0; nt < 8; ++nt) {
      v8f vacc[3]; for (int t = 0; t < 3; ++t) vacc[t] = zero8();
#pragma unroll
      for (int kt = 0; kt < 4; ++kt) {
        v16bf b = load_B(pack + OFF_WV, kt * 8 + nt, lane);
#pragma unroll
        for (int t = 0; t < 3; ++t) {
          v16bf a = load_A_lds(sX + t * 2048, 128, kt * 32, lane);
          vacc[t] = __builtin_amdgcn_wmma_f32_16x16x32_bf16(
              false, a, false, b, (short)0, vacc[t], false, false);
        }
      }
      int col = nt * 16 + l15;
#pragma unroll
      for (int v = 0; v < 8; ++v) {
        float f = av[0][v] * vacc[0][v] + av[1][v] * vacc[1][v] + av[2][v] * vacc[2][v];
        sT[(v + 8 * hi) * 128 + col] = (__bf16)f;
      }
    }
  }
  __syncthreads();

  // ---- feat = feat_pre @ Wo; conservative-head dot done in-register --------
  {
    const int kb[4] = {0, 32, 64, 96};
    v8f oacc[8]; for (int i = 0; i < 8; ++i) oacc[i] = zero8();
    gemm_acc<8, 4>(sT, 128, kb, pack + OFF_WO, oacc, lane);

    // cons partial: p[v] = sum_nt feat[row, nt*16+l15] * cons_W[nt*16+l15]
    float p[8];
#pragma unroll
    for (int v = 0; v < 8; ++v) p[v] = 0.f;
#pragma unroll
    for (int nt = 0; nt < 8; ++nt) {
      float w = cons_W[nt * 16 + l15];
#pragma unroll
      for (int v = 0; v < 8; ++v) p[v] += oacc[nt][v] * w;
    }
#pragma unroll
    for (int v = 0; v < 8; ++v) {
      float s = red16(p[v]);
      if (l15 == 0) sCons[v + 8 * hi] = s;
    }

    __syncthreads();
#pragma unroll
    for (int nt = 0; nt < 8; ++nt) {
      int col = nt * 16 + l15;
#pragma unroll
      for (int v = 0; v < 8; ++v) sX[(v + 8 * hi) * 128 + col] = (__bf16)oacc[nt][v];
    }
  }
  __syncthreads();

  // ---- form head: form_cov = feat @ spec(form_W) + form_b; dot with control
  {
    const int kb[4] = {0, 32, 64, 96};
    v8f facc[4]; for (int i = 0; i < 4; ++i) facc[i] = zero8();
    gemm_acc<4, 4>(sX, 128, kb, pack + OFF_FORM, facc, lane);
    float invf = scal[1];
    float p[8];
#pragma unroll
    for (int v = 0; v < 8; ++v) p[v] = 0.f;
#pragma unroll
    for (int nt = 0; nt < 4; ++nt) {
      int col = nt * 16 + l15;
      float fb = form_b[col];
#pragma unroll
      for (int v = 0; v < 8; ++v) {
        float fc = facc[nt][v] * invf + fb;
        float c  = sCTL[(v + 8 * hi) * 64 + col];
        p[v] += fc * c;
      }
    }
#pragma unroll
    for (int v = 0; v < 8; ++v) {
      float s = red16(p[v]);
      if (l15 == 0) sNC[v + 8 * hi] = s;
    }
  }
  __syncthreads();

  // ---- output --------------------------------------------------------------
  if (lane < 16) {
    out[row0 + lane] = sCons[lane] * scal[0] + cons_b[0] + sNC[lane];
  }
}

// ---------------------------------------------------------------------------
extern "C" void kernel_launch(void* const* d_in, const int* in_sizes, int n_in,
                              void* d_out, int out_size, void* d_ws, size_t ws_size,
                              hipStream_t stream) {
  const float* z       = (const float*)d_in[0];
  const float* action  = (const float*)d_in[1];
  const float* rw      = (const float*)d_in[2];
  const float* control = (const float*)d_in[3];
  const float* act_Wx  = (const float*)d_in[4];
  const float* act_bx  = (const float*)d_in[5];
  const float* act_Wz  = (const float*)d_in[6];
  const float* ctl_Wx  = (const float*)d_in[7];
  const float* ctl_bx  = (const float*)d_in[8];
  const float* ctl_Wz  = (const float*)d_in[9];
  const float* cht_Wx  = (const float*)d_in[10];
  const float* cht_bx  = (const float*)d_in[11];
  const float* cht_Wz  = (const float*)d_in[12];
  const float* We      = (const float*)d_in[13];
  const float* be      = (const float*)d_in[14];
  const float* Wq      = (const float*)d_in[15];
  const float* Wk      = (const float*)d_in[16];
  const float* Wv      = (const float*)d_in[17];
  const float* Wo      = (const float*)d_in[18];
  const float* cons_W  = (const float*)d_in[19];
  const float* cons_b  = (const float*)d_in[20];
  const float* form_W  = (const float*)d_in[21];
  const float* form_b  = (const float*)d_in[22];

  float*  scal = (float*)d_ws;
  __bf16* pack = (__bf16*)((char*)d_ws + 256);

  auto P = [&](const float* W, size_t off, int K, int N, int KT, int NT) {
    int total = KT * NT * 512;
    pack_kernel<<<(total + 255) / 256, 256, 0, stream>>>(W, pack + off, K, N, KT, NT);
  };
  P(act_Wx, OFF_ACT_WX, 96, 128, 3, 8);
  P(ctl_Wx, OFF_CTL_WX, 128, 128, 4, 8);
  P(cht_Wx, OFF_CHT_WX, 80, 128, 3, 8);
  P(act_Wz, OFF_ACT_WZ, 32, 64, 1, 4);
  P(ctl_Wz, OFF_CTL_WZ, 64, 64, 2, 4);
  P(cht_Wz, OFF_CHT_WZ, 16, 64, 1, 4);
  P(We,     OFF_WE,     64, 128, 2, 8);
  P(Wq,     OFF_WQ,     128, 128, 4, 8);
  P(Wk,     OFF_WK,     128, 128, 4, 8);
  P(Wv,     OFF_WV,     128, 128, 4, 8);
  P(Wo,     OFF_WO,     128, 128, 4, 8);
  P(form_W, OFF_FORM,   128, 64, 4, 4);

  spec_kernel<<<1, 128, 0, stream>>>(form_W, cons_W, scal);

  int B = in_sizes[0] / 64;
  reward_head_kernel<<<B / 16, 32, 0, stream>>>(
      z, action, rw, control, act_bx, ctl_bx, cht_bx, be,
      cons_W, cons_b, form_b, pack, scal, (float*)d_out);
}
